// DynamicMRConv4d_6631429505550
// MI455X (gfx1250) — matrix operand order repaired and checked by reading, not compiled, not run on hardware
//
#include <hip/hip_runtime.h>
#include <hip/hip_bf16.h>
#include <math.h>

#define BB 32
#define CC 192
#define HH 64
#define WW 64
#define HW 4096          // HH*WW
#define OO 384
#define KDIM 384         // 2*CC
#define NPIX 131072      // BB*HW

typedef __attribute__((ext_vector_type(16))) __bf16 v16bf;
typedef __attribute__((ext_vector_type(8)))  float  v8f;

// ---------------- kernel 1: norm = sum_c |x - roll(x,(32,32),(H,W))| ----------------
__global__ __launch_bounds__(256) void norm_kernel(const float* __restrict__ x,
                                                   float* __restrict__ normbuf) {
  int p = blockIdx.x * 256 + threadIdx.x;
  int b = p >> 12, hw = p & 4095;
  int h = hw >> 6, w = hw & 63;
  int hw2 = (((h + 32) & 63) << 6) | ((w + 32) & 63);
  const float* base = x + (size_t)b * CC * HW;
  float s = 0.f;
  #pragma unroll 4
  for (int c = 0; c < CC; ++c) {
    const float* cb = base + (size_t)c * HW;
    s += fabsf(cb[hw] - cb[hw2]);
  }
  normbuf[p] = s;
}

// ---------------- kernel 2: thr[b] = mean - unbiased std over H*W ----------------
__global__ __launch_bounds__(256) void stats_kernel(const float* __restrict__ normbuf,
                                                    float* __restrict__ thr) {
  __shared__ float ssum[256], ssq[256];
  int b = blockIdx.x, t = threadIdx.x;
  const float* nb = normbuf + b * HW;
  float s = 0.f, q = 0.f;
  for (int i = t; i < HW; i += 256) { float v = nb[i]; s += v; q += v * v; }
  ssum[t] = s; ssq[t] = q;
  __syncthreads();
  for (int o = 128; o > 0; o >>= 1) {
    if (t < o) { ssum[t] += ssum[t + o]; ssq[t] += ssq[t + o]; }
    __syncthreads();
  }
  if (t == 0) {
    float mu  = ssum[0] / (float)HW;
    float var = (ssq[0] - (float)HW * mu * mu) / (float)(HW - 1);  // ddof=1
    thr[b] = mu - sqrtf(fmaxf(var, 0.f));
  }
}

// ---------------- kernel 3: 16-bit shift mask per pixel ----------------
__global__ __launch_bounds__(256) void mask_kernel(const float* __restrict__ x,
                                                   const float* __restrict__ thr,
                                                   unsigned* __restrict__ maskbuf) {
  int p = blockIdx.x * 256 + threadIdx.x;
  int b = p >> 12, hw = p & 4095;
  int h = hw >> 6, w = hw & 63;
  const float* base = x + (size_t)b * CC * HW;
  int offV[8], offH[8];
  offV[0] = hw; offH[0] = hw;
  #pragma unroll
  for (int i = 1; i < 8; ++i) {
    offV[i] = (((h - 8 * i) & 63) << 6) | w;   // roll(+s) axis 2: r[h] = x[h-s]
    offH[i] = (h << 6) | ((w - 8 * i) & 63);   // roll(+s) axis 3
  }
  float dv[8] = {0.f}, dh[8] = {0.f};
  for (int c = 0; c < CC; ++c) {
    const float* cb = base + (size_t)c * HW;
    float xc = cb[hw];
    #pragma unroll
    for (int i = 1; i < 8; ++i) dv[i] += fabsf(xc - cb[offV[i]]);
    #pragma unroll
    for (int i = 1; i < 8; ++i) dh[i] += fabsf(xc - cb[offH[i]]);
  }
  float t = thr[b];
  unsigned m = 0;
  #pragma unroll
  for (int i = 0; i < 8; ++i) {
    if (dv[i] < t) m |= 1u << i;
    if (dh[i] < t) m |= 1u << (8 + i);
  }
  maskbuf[p] = m;
}

// ---------------- kernel 4: x_j = max over active shifts of (r - x), >= 0 ----------------
__global__ __launch_bounds__(256) void xj_kernel(const float* __restrict__ x,
                                                 const unsigned* __restrict__ maskbuf,
                                                 float* __restrict__ xj) {
  int p = blockIdx.x * 256 + threadIdx.x;
  int b = p >> 12, hw = p & 4095;
  int h = hw >> 6, w = hw & 63;
  unsigned m = maskbuf[p] & 0xFEFEu;   // s=0 shifts are no-ops under max(.,0)
  const float* base = x + (size_t)b * CC * HW;
  float* xb = xj + (size_t)b * CC * HW;
  int offV[8], offH[8];
  #pragma unroll
  for (int i = 1; i < 8; ++i) {
    offV[i] = (((h - 8 * i) & 63) << 6) | w;
    offH[i] = (h << 6) | ((w - 8 * i) & 63);
  }
  for (int c = 0; c < CC; ++c) {
    const float* cb = base + (size_t)c * HW;
    float xc = cb[hw];
    float v = 0.f;
    #pragma unroll
    for (int i = 1; i < 8; ++i) {
      if (m & (1u << i))        v = fmaxf(v, cb[offV[i]] - xc);
      if (m & (1u << (8 + i)))  v = fmaxf(v, cb[offH[i]] - xc);
    }
    xb[(size_t)c * HW + hw] = v;
  }
}

// ---------------- kernel 5: bf16-split WMMA GEMM  y = W * [x;x_j] + bias ----------------
// fp32 emulated as hi+lo bf16:  A*B ~= Ahi*Bhi + Ahi*Blo + Alo*Bhi  (fp32 accum).
// Block tile 128(M) x 128(N); K-chunks of 32; 8 waves, each 64(M) x 32(N).
#define TM 128
#define TN 128
#define TKC 32
#define LDK 40           // bf16 row stride: multiple of 8 -> every 8-elem run 16B aligned

union Frag16 { uint4 u[2]; v16bf v; };

static __device__ __forceinline__ void cvt_split_store(__bf16* hp, __bf16* lp, float4 v) {
  __bf16 h0 = (__bf16)v.x, h1 = (__bf16)v.y, h2 = (__bf16)v.z, h3 = (__bf16)v.w;
  __bf16 l0 = (__bf16)(v.x - (float)h0), l1 = (__bf16)(v.y - (float)h1);
  __bf16 l2 = (__bf16)(v.z - (float)h2), l3 = (__bf16)(v.w - (float)h3);
  union { __bf16 b[4]; unsigned long long u; } H, L;
  H.b[0] = h0; H.b[1] = h1; H.b[2] = h2; H.b[3] = h3;
  L.b[0] = l0; L.b[1] = l1; L.b[2] = l2; L.b[3] = l3;
  *(unsigned long long*)hp = H.u;
  *(unsigned long long*)lp = L.u;
}

__global__ __launch_bounds__(256) void gemm_kernel(const float* __restrict__ cw,
                                                   const float* __restrict__ cb,
                                                   const float* __restrict__ x,
                                                   const float* __restrict__ xj,
                                                   float* __restrict__ y) {
  __shared__ __bf16 Ahi[TM][LDK], Alo[TM][LDK];   // A[m][k]
  __shared__ __bf16 Bhi[TN][LDK], Blo[TN][LDK];   // B transposed: [n][k]
  const int tid = threadIdx.x;
  const int wid = tid >> 5, lane = tid & 31;
  const int l15 = lane & 15, grp = lane >> 4;
  const int mBase = blockIdx.y * TM;
  const int nBase = blockIdx.x * TN;
  const int b = nBase >> 12, hw0 = nBase & 4095;   // N-tile stays inside one image
  const int waveM = (wid >> 2) * 64, waveN = (wid & 3) * 32;

  // staging thread mappings
  const int sm = tid >> 1, skh = (tid & 1) << 4;      // A: row m, k-half
  const int sn = tid & 127, skq = (tid >> 7) << 4;    // B: col n, k-half

  v8f acc[4][2];
  #pragma unroll
  for (int i = 0; i < 4; ++i)
    #pragma unroll
    for (int j = 0; j < 2; ++j) acc[i][j] = (v8f){};

  for (int kk = 0; kk < KDIM; kk += TKC) {
    // ---- stage A: 128x32 fp32 -> hi/lo bf16; thread owns (m=sm, k=skh..skh+15)
    {
      const float* arow = cw + (size_t)(mBase + sm) * KDIM + kk + skh;
      #pragma unroll
      for (int i = 0; i < 4; ++i) {
        float4 v = *(const float4*)(arow + 4 * i);
        cvt_split_store(&Ahi[sm][skh + 4 * i], &Alo[sm][skh + 4 * i], v);
      }
    }
    // ---- stage B: 32(k) x 128(n); thread owns (n=sn, k=skq..skq+15)
    // chunk is entirely in x (kk<192) or entirely in xj (kk>=192) since 192%32==0
    {
      const float* sb = ((kk < CC) ? (x  + (size_t)(b * CC + kk) * HW)
                                   : (xj + (size_t)(b * CC + kk - CC) * HW)) + hw0 + sn;
      #pragma unroll
      for (int i = 0; i < 4; ++i) {
        float4 v;
        v.x = sb[(size_t)(skq + 4 * i + 0) * HW];
        v.y = sb[(size_t)(skq + 4 * i + 1) * HW];
        v.z = sb[(size_t)(skq + 4 * i + 2) * HW];
        v.w = sb[(size_t)(skq + 4 * i + 3) * HW];
        cvt_split_store(&Bhi[sn][skq + 4 * i], &Blo[sn][skq + 4 * i], v);
      }
    }
    __syncthreads();

    // ---- B fragments (both n-subtiles, hi+lo): lane n=l15, K = 16*grp .. 16*grp+15
    Frag16 bh[2], bl[2];
    #pragma unroll
    for (int nt = 0; nt < 2; ++nt) {
      const __bf16* ph = &Bhi[waveN + nt * 16 + l15][16 * grp];
      const __bf16* pl = &Blo[waveN + nt * 16 + l15][16 * grp];
      bh[nt].u[0] = *(const uint4*)(ph);     bh[nt].u[1] = *(const uint4*)(ph + 8);
      bl[nt].u[0] = *(const uint4*)(pl);     bl[nt].u[1] = *(const uint4*)(pl + 8);
    }
    // ---- per m-subtile: A fragment (lane m=l15, K = 16g..16g+15)
    #pragma unroll
    for (int mt = 0; mt < 4; ++mt) {
      const __bf16* ph = &Ahi[waveM + mt * 16 + l15][0];
      const __bf16* pl = &Alo[waveM + mt * 16 + l15][0];
      Frag16 ah, al;
      ah.u[0] = *(const uint4*)(ph + 16 * grp);
      ah.u[1] = *(const uint4*)(ph + 16 * grp + 8);
      al.u[0] = *(const uint4*)(pl + 16 * grp);
      al.u[1] = *(const uint4*)(pl + 16 * grp + 8);
      #pragma unroll
      for (int nt = 0; nt < 2; ++nt) {
        acc[mt][nt] = __builtin_amdgcn_wmma_f32_16x16x32_bf16(
            false, ah.v, false, bh[nt].v, (short)0, acc[mt][nt], false, false);
        acc[mt][nt] = __builtin_amdgcn_wmma_f32_16x16x32_bf16(
            false, ah.v, false, bl[nt].v, (short)0, acc[mt][nt], false, false);
        acc[mt][nt] = __builtin_amdgcn_wmma_f32_16x16x32_bf16(
            false, al.v, false, bh[nt].v, (short)0, acc[mt][nt], false, false);
      }
    }
    __syncthreads();
  }

  // epilogue: +bias, write y into d_out [b, o, hw]
  float* outb = y + (size_t)b * OO * HW + hw0;
  #pragma unroll
  for (int mt = 0; mt < 4; ++mt) {
    #pragma unroll
    for (int nt = 0; nt < 2; ++nt) {
      int nloc = waveN + nt * 16 + l15;
      #pragma unroll
      for (int j = 0; j < 8; ++j) {
        int m = mBase + waveM + mt * 16 + grp * 8 + j;  // C/D: VGPR j, lane-group -> M
        outb[(size_t)m * HW + nloc] = acc[mt][nt][j] + cb[m];
      }
    }
  }
}

// ---------------- kernel 6: per-channel BN scale/shift (batch stats, biased var) ----------------
__global__ __launch_bounds__(256) void bnstat_kernel(const float* __restrict__ y,
                                                     const float* __restrict__ gamma,
                                                     const float* __restrict__ beta,
                                                     float* __restrict__ scale,
                                                     float* __restrict__ shift) {
  __shared__ float ssum[256], ssq[256];
  int o = blockIdx.x, t = threadIdx.x;
  float s = 0.f, q = 0.f;
  for (int b = 0; b < BB; ++b) {
    const float* yb = y + ((size_t)b * OO + o) * HW;
    for (int i = t; i < HW; i += 256) { float v = yb[i]; s += v; q += v * v; }
  }
  ssum[t] = s; ssq[t] = q;
  __syncthreads();
  for (int off = 128; off > 0; off >>= 1) {
    if (t < off) { ssum[t] += ssum[t + off]; ssq[t] += ssq[t + off]; }
    __syncthreads();
  }
  if (t == 0) {
    float inv = 1.f / (float)(BB * HW);
    float mu  = ssum[0] * inv;
    float var = ssq[0] * inv - mu * mu;                 // biased (training-mode forward)
    float sc  = gamma[o] * rsqrtf(var + 1e-5f);
    scale[o] = sc;
    shift[o] = beta[o] - mu * sc;
  }
}

// ---------------- kernel 7: in-place BN + exact-erf GELU ----------------
__global__ __launch_bounds__(256) void bngelu_kernel(float* __restrict__ y,
                                                     const float* __restrict__ scale,
                                                     const float* __restrict__ shift) {
  size_t i = ((size_t)blockIdx.x * 256 + threadIdx.x) * 4;
  int o = (int)((i >> 12) % OO);                        // layout (b*OO+o)*HW, HW%4==0
  float sc = scale[o], sh = shift[o];
  float4 v = *(float4*)(y + i);
  float4 r;
  float u;
  u = v.x * sc + sh; r.x = 0.5f * u * (1.f + erff(u * 0.70710678118654752f));
  u = v.y * sc + sh; r.y = 0.5f * u * (1.f + erff(u * 0.70710678118654752f));
  u = v.z * sc + sh; r.z = 0.5f * u * (1.f + erff(u * 0.70710678118654752f));
  u = v.w * sc + sh; r.w = 0.5f * u * (1.f + erff(u * 0.70710678118654752f));
  *(float4*)(y + i) = r;
}

extern "C" void kernel_launch(void* const* d_in, const int* in_sizes, int n_in,
                              void* d_out, int out_size, void* d_ws, size_t ws_size,
                              hipStream_t stream) {
  const float* x     = (const float*)d_in[0];
  const float* cw    = (const float*)d_in[1];
  const float* cb    = (const float*)d_in[2];
  const float* gamma = (const float*)d_in[3];
  const float* beta  = (const float*)d_in[4];
  float* out = (float*)d_out;

  // workspace layout (floats)
  float*    ws      = (float*)d_ws;
  float*    thr     = ws;                       // 32
  float*    bnscale = ws + 32;                  // 384
  float*    bnshift = ws + 32 + OO;             // 384
  float*    normbuf = ws + 1024;                // NPIX
  unsigned* maskbuf = (unsigned*)(ws + 1024 + NPIX);        // NPIX u32
  float*    xj      = ws + 1024 + 2 * NPIX;     // BB*CC*HW = 25.2M floats

  norm_kernel <<<NPIX / 256, 256, 0, stream>>>(x, normbuf);
  stats_kernel<<<BB,         256, 0, stream>>>(normbuf, thr);
  mask_kernel <<<NPIX / 256, 256, 0, stream>>>(x, thr, maskbuf);
  xj_kernel   <<<NPIX / 256, 256, 0, stream>>>(x, maskbuf, xj);
  gemm_kernel <<<dim3(NPIX / TN, OO / TM), 256, 0, stream>>>(cw, cb, x, xj, out);
  bnstat_kernel<<<OO,        256, 0, stream>>>(out, gamma, beta, bnscale, bnshift);
  bngelu_kernel<<<(unsigned)(((size_t)BB * OO * HW) / (256 * 4)), 256, 0, stream>>>(out, bnscale, bnshift);
}
